// VAENF_2207613190724
// MI455X (gfx1250) — matrix operand order repaired and checked by reading, not compiled, not run on hardware
//
#include <hip/hip_runtime.h>
#include <hip/hip_bf16.h>
#include <math.h>

// Problem constants (match reference)
#define B_SZ   16384
#define DIM_   4096
#define NL_    64
#define NH_    20
#define K_FLOW 8
#define NP_    (NH_ * NH_ + 2 * NH_)   // 440
#define NPK_   (NP_ * K_FLOW)          // 3520

typedef __attribute__((ext_vector_type(16))) __bf16 v16bf;
typedef __attribute__((ext_vector_type(8)))  float  v8f;
typedef __attribute__((ext_vector_type(4)))  float  v4f;

// ---------------------------------------------------------------------------
// Fragment loaders: f32 global memory -> bf16 WMMA fragments.
// A-matrix (16x32, 16-bit): lanes 0-15 hold M=0..15 / K base 0, lanes 16-31
// hold same M rows / K base +8; element e in [0,8) -> K = k+e (contiguous),
// e in [8,16) -> K = k+16+e-8 (second contiguous run). Two v4f per run.
// B-matrix (32x16): lane owns column N = lane&15; same K runs as A
// (K base +8 for lanes 16-31, second run at +16) -> two 8-float runs per lane.
// ---------------------------------------------------------------------------
template<int E0>
__device__ inline void load_run8(v16bf& v, const float* __restrict__ row, int k, int KD) {
    if (k + 8 <= KD) {
        const v4f* p = reinterpret_cast<const v4f*>(row + k);
        v4f f0 = p[0], f1 = p[1];
#pragma unroll
        for (int e = 0; e < 4; ++e) v[E0 + e]     = (__bf16)f0[e];
#pragma unroll
        for (int e = 0; e < 4; ++e) v[E0 + 4 + e] = (__bf16)f1[e];
    } else {
#pragma unroll
        for (int e = 0; e < 8; ++e) {
            float f = (k + e < KD) ? row[k + e] : 0.0f;
            v[E0 + e] = (__bf16)f;
        }
    }
}

__device__ inline void load_run16(v16bf& v, const float* __restrict__ row, int k, int KD) {
    if (k + 16 <= KD) {
        const v4f* p = reinterpret_cast<const v4f*>(row + k);
#pragma unroll
        for (int q = 0; q < 4; ++q) {
            v4f f = p[q];
#pragma unroll
            for (int e = 0; e < 4; ++e) v[q * 4 + e] = (__bf16)f[e];
        }
    } else {
#pragma unroll
        for (int e = 0; e < 16; ++e) {
            float f = (k + e < KD) ? row[k + e] : 0.0f;
            v[e] = (__bf16)f;
        }
    }
}

// ---------------------------------------------------------------------------
// Generic WMMA GEMM: out[M,N] = act(A[M,KD] @ W[N,KD]^T + bias[N])
// 256 threads = 8 waves. Each wave owns NT_TILES consecutive 16x16 output
// tiles sharing one A fragment per K-step (5x/4x A-fragment register reuse,
// and NT_TILES independent accumulators -> no WMMA->WMMA RAW hazards).
// Block covers 128 rows x (16*NT_TILES) cols.
// NTOUT: nontemporal stores for stream-once outputs (recon, 268 MB) so the
// hot L2 working set (weights + h + hd, all << 192 MB) isn't evicted.
// ---------------------------------------------------------------------------
template<int KD, int NT_TILES, bool RELU, bool COLSUM, bool NTOUT>
__global__ __launch_bounds__(256)
void gemm_wmma_kernel(const float* __restrict__ A, const float* __restrict__ W,
                      const float* __restrict__ bias, float* __restrict__ out,
                      int N, float* __restrict__ colsum) {
    const int lane = threadIdx.x & 31;
    const int wave = threadIdx.x >> 5;
    const int hi   = lane >> 4;       // half-wave selector
    const int ln   = lane & 15;
    const int m0   = (blockIdx.x * 8 + wave) * 16;
    const int n0   = blockIdx.y * (16 * NT_TILES);

    const float* arow  = A + (size_t)(m0 + ln) * KD;
    const float* wbase = W + (size_t)(n0 + ln) * KD;

    v8f acc[NT_TILES] = {};
    const int NK = (KD + 31) / 32;
    for (int kt = 0; kt < NK; ++kt) {
        const int k0 = kt * 32;
        v16bf af;
        load_run8<0>(af, arow, k0 + hi * 8, KD);
        load_run8<8>(af, arow, k0 + 16 + hi * 8, KD);
#pragma unroll
        for (int t = 0; t < NT_TILES; ++t) {
            const float* wrow = wbase + (size_t)(16 * t) * KD;
            v16bf bfm;
            load_run8<0>(bfm, wrow, k0 + hi * 8, KD);
            load_run8<8>(bfm, wrow, k0 + 16 + hi * 8, KD);
            acc[t] = __builtin_amdgcn_wmma_f32_16x16x32_bf16(
                /*neg_a=*/false, af, /*neg_b=*/false, bfm,
                /*c_mod=*/(short)0, acc[t], /*reuse_a=*/false, /*reuse_b=*/false);
        }
    }

    // C/D layout: VGPR r -> row m0 + (hi?8:0) + r, col (tile base) + ln
#pragma unroll
    for (int t = 0; t < NT_TILES; ++t) {
        const int col = n0 + 16 * t + ln;
        const float bv = bias[col];
        float csum = 0.0f;
#pragma unroll
        for (int r = 0; r < 8; ++r) {
            const int row = m0 + hi * 8 + r;
            float val = acc[t][r] + bv;
            if (RELU) val = fmaxf(val, 0.0f);
            float* dst = out + (size_t)row * N + col;
            if (NTOUT) __builtin_nontemporal_store(val, dst);
            else       *dst = val;
            if (COLSUM) csum += val;
        }
        if (COLSUM) {
            csum += __shfl_xor(csum, 16, 32);   // lane L + L^16 share a column
            if (hi == 0) atomicAdd(&colsum[col], csum);
        }
    }
}

// ---------------------------------------------------------------------------
// Small helper kernels
// ---------------------------------------------------------------------------
__global__ void zero_kernel(float* p, int n) {
    int i = blockIdx.x * blockDim.x + threadIdx.x;
    if (i < n) p[i] = 0.0f;
}

// flow_p = (hsum / B) @ W23^T + b23   (mean commutes with the linear map)
__global__ void flowp_kernel(const float* __restrict__ hsum,
                             const float* __restrict__ W23,
                             const float* __restrict__ b23,
                             float* __restrict__ flowp) {
    int n = blockIdx.x * blockDim.x + threadIdx.x;
    if (n >= NPK_) return;
    const float* wr = W23 + (size_t)n * 400;
    float s = 0.0f;
    for (int i = 0; i < 400; ++i) s += hsum[i] * wr[i];
    flowp[n] = s * (1.0f / (float)B_SZ) + b23[n];
}

// z = eps * exp(0.5 * logvar) + mu   (elementwise over B*NL)
__global__ void z_kernel(const float* __restrict__ eps,
                         const float* __restrict__ mu,
                         const float* __restrict__ lv,
                         float* __restrict__ z, int n) {
    int i = blockIdx.x * blockDim.x + threadIdx.x;
    if (i < n) z[i] = eps[i] * expf(0.5f * lv[i]) + mu[i];
}

// K=8 sequential TriagSylvester flow steps; one lane per batch row; only the
// first NH=20 coordinates of z mutate. Flow params staged in LDS (14 KB).
__global__ __launch_bounds__(256)
void flow_kernel(float* __restrict__ z, const float* __restrict__ flowp) {
    __shared__ float P[NPK_];
    for (int i = threadIdx.x; i < NPK_; i += blockDim.x) P[i] = flowp[i];
    __syncthreads();

    const int row = blockIdx.x * blockDim.x + threadIdx.x;
    float* zp = z + (size_t)row * NL_;
    float zr[NH_];
#pragma unroll
    for (int i = 0; i < NH_; ++i) zr[i] = zp[i];

    for (int k = 0; k < K_FLOW; ++k) {
        const float* p = P + k * NP_;
        const float* v = p + NH_ * NH_;        // diag replacement
        const float* b = p + NH_ * NH_ + NH_;  // bias
        const bool permute = ((k & 1) == 0);

        float t[NH_];
        for (int j = 0; j < NH_; ++j) {
            float pre = b[j];
#pragma unroll
            for (int i = 0; i < NH_; ++i) {
                float w;
                if (i == j)        w = v[j];
                else if (permute)  w = (j > i) ? p[j * NH_ + i] : 0.0f;
                else               w = (i > j) ? p[i * NH_ + j] : 0.0f;
                pre += w * zr[i];
            }
            t[j] = tanhf(pre);
        }
        for (int j = 0; j < NH_; ++j) {
            float u = 0.0f;
#pragma unroll
            for (int i = 0; i < NH_; ++i) {
                float r;
                if (permute) r = (j >= i) ? p[i * NH_ + j] : 0.0f;
                else         r = (i >= j) ? p[j * NH_ + i] : 0.0f;
                u += r * t[i];
            }
            zr[j] += u;
        }
    }
#pragma unroll
    for (int i = 0; i < NH_; ++i) zp[i] = zr[i];
}

// ---------------------------------------------------------------------------
// Launcher. Workspace layout (f32), total ~55 MB:
//   [0)        hsum   : 400
//   [4096)     flowp  : 3520
//   [32768)    h      : B*400
//   [+h]       z      : B*64
//   [+z]       hd     : B*400
// d_out: recon (B*DIM) | mu (B*NL) | logvar (B*NL)
// ---------------------------------------------------------------------------
extern "C" void kernel_launch(void* const* d_in, const int* in_sizes, int n_in,
                              void* d_out, int out_size, void* d_ws, size_t ws_size,
                              hipStream_t stream) {
    const float* x   = (const float*)d_in[0];
    const float* eps = (const float*)d_in[1];
    const float* W1  = (const float*)d_in[2];
    const float* b1  = (const float*)d_in[3];
    const float* W21 = (const float*)d_in[4];
    const float* b21 = (const float*)d_in[5];
    const float* W22 = (const float*)d_in[6];
    const float* b22 = (const float*)d_in[7];
    const float* W23 = (const float*)d_in[8];
    const float* b23 = (const float*)d_in[9];
    const float* W3  = (const float*)d_in[10];
    const float* b3  = (const float*)d_in[11];
    const float* W4  = (const float*)d_in[12];
    const float* b4  = (const float*)d_in[13];

    float* out   = (float*)d_out;
    float* recon = out;
    float* mu    = out + (size_t)B_SZ * DIM_;
    float* lv    = mu + (size_t)B_SZ * NL_;

    char*  ws    = (char*)d_ws;
    float* hsum  = (float*)(ws);
    float* flowp = (float*)(ws + 4096);
    float* h     = (float*)(ws + 32768);
    float* z     = h + (size_t)B_SZ * 400;
    float* hd    = z + (size_t)B_SZ * NL_;

    // 1) zero column-sum accumulator (ws is poisoned, not re-poisoned between replays)
    zero_kernel<<<1, 512, 0, stream>>>(hsum, 400);

    // 2) h = relu(x @ W1^T + b1), fused column sums for batch mean
    //    N=400 -> 5 tiles/wave, grid.y = 400/80 = 5
    {
        dim3 grid(B_SZ / 128, 5);
        gemm_wmma_kernel<DIM_, 5, true, true, false>
            <<<grid, 256, 0, stream>>>(x, W1, b1, h, 400, hsum);
    }

    // 3) flow_p = (mean_B h) @ W23^T + b23
    flowp_kernel<<<(NPK_ + 255) / 256, 256, 0, stream>>>(hsum, W23, b23, flowp);

    // 4,5) mu / logvar directly into d_out  (N=64 -> 4 tiles/wave, grid.y=1)
    {
        dim3 grid(B_SZ / 128, 1);
        gemm_wmma_kernel<400, 4, false, false, false>
            <<<grid, 256, 0, stream>>>(h, W21, b21, mu, NL_, nullptr);
        gemm_wmma_kernel<400, 4, false, false, false>
            <<<grid, 256, 0, stream>>>(h, W22, b22, lv, NL_, nullptr);
    }

    // 6) reparameterize
    z_kernel<<<(B_SZ * NL_) / 256, 256, 0, stream>>>(eps, mu, lv, z, B_SZ * NL_);

    // 7) 8 flow steps in place on z[:, :20]
    flow_kernel<<<B_SZ / 256, 256, 0, stream>>>(z, flowp);

    // 8) hd = relu(z @ W3^T + b3)  (N=400 -> 5 tiles/wave, grid.y=5)
    {
        dim3 grid(B_SZ / 128, 5);
        gemm_wmma_kernel<NL_, 5, true, false, false>
            <<<grid, 256, 0, stream>>>(z, W3, b3, hd, 400, nullptr);
    }

    // 9) recon = hd @ W4^T + b4 into d_out, nontemporal stores (write-once 268 MB)
    //    N=4096 -> 4 tiles/wave, grid.y = 4096/64 = 64
    {
        dim3 grid(B_SZ / 128, 64);
        gemm_wmma_kernel<400, 4, false, false, true>
            <<<grid, 256, 0, stream>>>(hd, W4, b4, recon, DIM_, nullptr);
    }
}
